// TimeAwareMultiHeadAttention_9088150798559
// MI455X (gfx1250) — matrix operand order repaired and checked by reading, not compiled, not run on hardware
//
#include <hip/hip_runtime.h>

// ---------------------------------------------------------------------------
// Time-aware MHA for MI455X (gfx1250, wave32, WMMA).
//   B=2, S=2048, D=1024, H=16, dk=64.
// Pipeline:
//   gemm128<1>: Q = q @ Wq^T + bq   -> f16 [B,H,S,dk]
//   gemm128<1>: K = k @ Wk^T + bk   -> f16 [B,H,S,dk]
//   gemm128<2>: V = v @ Wv^T + bv   -> f16 [B,H,dk,S]  (transposed)
//   attn_kernel: flash-style online softmax w/ exp(-|dt|) weighting + mask,
//                K/V tiles staged via GLOBAL_LOAD_ASYNC_TO_LDS (ASYNCcnt)
//                -> f32 ctx [B,S,D]
//   gemm128<0>: out = ctx @ Wo^T + bo -> f32 [B,S,D]
// Workspace: 3 * 8MB (f16 QKV) + 16MB (f32 ctx) = 40MB.
// ---------------------------------------------------------------------------

#define USE_ASYNC_LDS 1   // gfx1250 async global->LDS staging in attention

typedef _Float16 v16h __attribute__((ext_vector_type(16)));
typedef _Float16 v8h  __attribute__((ext_vector_type(8)));
typedef _Float16 v4h  __attribute__((ext_vector_type(4)));
typedef float    v8f  __attribute__((ext_vector_type(8)));
typedef float    f4   __attribute__((ext_vector_type(4)));

union V16U { v16h v; v8h h[2]; };

#define D_MODEL 1024
#define NHEADS  16
#define DKH     64
#define BATCH   2
#define SEQ     2048
#define MROWS   (BATCH * SEQ)

static __device__ __forceinline__ v8f wmma_f16(v16h a, v16h b, v8f c) {
  // (neg_a, A, neg_b, B, c_mod, C, reuse_a, reuse_b)
  return __builtin_amdgcn_wmma_f32_16x16x32_f16(false, a, false, b, (short)0, c,
                                                false, false);
}

// ---------------------------------------------------------------------------
// C[M,N] = X[M,K] @ W[N,K]^T + bias.  M=4096, N=1024, K=1024.
// Block: 256 threads (8 waves), tile 128x128, BK=32.
// Wave (wm=w>>1 in 0..3, wn=w&1 in 0..1) owns rows 32*wm..+31, cols 64*wn..+63
// -> 2x4 = 8 WMMA 16x16 accumulators per wave.
// MODE 0: f32 row-major [M, D_MODEL]
// MODE 1: f16 [B,H,S,dk]   (Q, K)
// MODE 2: f16 [B,H,dk,S]   (V transposed)
// ---------------------------------------------------------------------------
template <int MODE>
__global__ __launch_bounds__(256) void gemm128(const float* __restrict__ X,
                                               const float* __restrict__ W,
                                               const float* __restrict__ bias,
                                               void* __restrict__ outp) {
  __shared__ __align__(16) _Float16 As[128 * 32];
  __shared__ __align__(16) _Float16 Bs[128 * 32];

  const int m0   = blockIdx.y * 128;
  const int n0   = blockIdx.x * 128;
  const int t    = threadIdx.x;
  const int wave = t >> 5;
  const int lane = t & 31;
  const int l16  = lane & 15;
  const int hsel = lane >> 4;   // half-wave select
  const int wm   = wave >> 1;
  const int wn   = wave & 1;

  v8f acc[2][4] = {};

  for (int k0 = 0; k0 < D_MODEL; k0 += 32) {
    __syncthreads();
    // Stage 128x32 tiles of X and W into LDS as f16 (f32->f16 on the fly),
    // and prefetch the next k-tile into cache (global_prefetch_b8).
#pragma unroll
    for (int i = 0; i < 4; i++) {
      int c   = i * 256 + t;          // 0..1023
      int row = c >> 3;               // 0..127
      int kc  = (c & 7) * 4;          // 0..28
      const float* xp = X + (size_t)(m0 + row) * D_MODEL + k0 + kc;
      const float* wp = W + (size_t)(n0 + row) * D_MODEL + k0 + kc;
      f4 xa = *(const f4*)xp;
      f4 wa = *(const f4*)wp;
      if (k0 + 32 < D_MODEL) {
        __builtin_prefetch(xp + 32, 0, 3);
        __builtin_prefetch(wp + 32, 0, 3);
      }
      *(v4h*)(As + row * 32 + kc) = __builtin_convertvector(xa, v4h);
      *(v4h*)(Bs + row * 32 + kc) = __builtin_convertvector(wa, v4h);
    }
    __syncthreads();

    V16U a[2], b[4];
#pragma unroll
    for (int rt = 0; rt < 2; rt++) {
      const _Float16* ap = As + (32 * wm + 16 * rt + l16) * 32;
      a[rt].h[0] = *(const v8h*)(ap + hsel * 8);         // K = sel*8..+7
      a[rt].h[1] = *(const v8h*)(ap + 16 + hsel * 8);    // K = 16+sel*8..+7
    }
#pragma unroll
    for (int ct = 0; ct < 4; ct++) {
      const _Float16* bp = Bs + (64 * wn + 16 * ct + l16) * 32;
      b[ct].h[0] = *(const v8h*)(bp + hsel * 8);         // K = sel*8..+7
      b[ct].h[1] = *(const v8h*)(bp + 16 + hsel * 8);    // K = 16+sel*8..+7
    }
#pragma unroll
    for (int rt = 0; rt < 2; rt++)
#pragma unroll
      for (int ct = 0; ct < 4; ct++)
        acc[rt][ct] = wmma_f16(a[rt].v, b[ct].v, acc[rt][ct]);
  }

  // Epilogue: bias add + layout-specific store.
#pragma unroll
  for (int ct = 0; ct < 4; ct++) {
    int gn = n0 + 64 * wn + 16 * ct + l16;
    float bvv = bias[gn];
#pragma unroll
    for (int rt = 0; rt < 2; rt++) {
#pragma unroll
      for (int v = 0; v < 8; v++) {
        int gm = m0 + 32 * wm + 16 * rt + v + 8 * hsel;
        float r = acc[rt][ct][v] + bvv;
        if (MODE == 0) {
          ((float*)outp)[(size_t)gm * D_MODEL + gn] = r;
        } else {
          int bb = gm >> 11;            // / SEQ
          int ss = gm & (SEQ - 1);
          int hh = gn >> 6;             // / DKH
          int dd = gn & (DKH - 1);
          if (MODE == 1)
            ((_Float16*)outp)[(((size_t)bb * NHEADS + hh) * SEQ + ss) * DKH + dd] =
                (_Float16)r;
          else
            ((_Float16*)outp)[(((size_t)bb * NHEADS + hh) * DKH + dd) * SEQ + ss] =
                (_Float16)r;
        }
      }
    }
  }
}

// ---------------------------------------------------------------------------
// Flash attention with time impact.  Grid: (SEQ/64, B*H), block 128 (4 waves).
// Each wave owns 16 queries; waves share 32-key K/V tiles staged in LDS.
// ---------------------------------------------------------------------------
__global__ __launch_bounds__(128) void attn_kernel(
    const _Float16* __restrict__ Qh, const _Float16* __restrict__ Kh,
    const _Float16* __restrict__ VTh, const float* __restrict__ tdiff,
    const unsigned char* __restrict__ maskp, float* __restrict__ ctx) {
  __shared__ __align__(16) _Float16 Kt[32 * 64];     // [key][d]
  __shared__ __align__(16) _Float16 Vt[64 * 32];     // [d][key]
  __shared__ __align__(16) _Float16 Pt[4][16 * 32];  // per-wave [q][key]

  const int bh   = blockIdx.y;
  const int bb   = bh / NHEADS;
  const int hh   = bh % NHEADS;
  const int t    = threadIdx.x;
  const int wave = t >> 5;
  const int lane = t & 31;
  const int l16  = lane & 15;
  const int hsel = lane >> 4;
  const int q0   = blockIdx.x * 64 + wave * 16;
  const float rscale = 0.125f;  // 1/sqrt(dk)
  const float NEGINF = -__builtin_inff();

  // Q A-fragments (d-chunks 0..31 and 32..63), loaded once.
  V16U aq[2];
  {
    const _Float16* qp = Qh + ((size_t)bh * SEQ + q0 + l16) * DKH;
#pragma unroll
    for (int c = 0; c < 2; c++) {
      aq[c].h[0] = *(const v8h*)(qp + 32 * c + hsel * 8);
      aq[c].h[1] = *(const v8h*)(qp + 32 * c + 16 + hsel * 8);
    }
  }

  float tq[8], mrow[8], lrow[8];
#pragma unroll
  for (int v = 0; v < 8; v++) {
    tq[v]   = tdiff[(size_t)bb * SEQ + q0 + v + 8 * hsel];
    mrow[v] = NEGINF;
    lrow[v] = 0.f;
  }
  v8f o[4] = {};

#if USE_ASYNC_LDS
  const unsigned long long kbase = (unsigned long long)(const void*)Kh;
  const unsigned long long vbase = (unsigned long long)(const void*)VTh;
#endif

  for (int kt = 0; kt < SEQ; kt += 32) {
    __syncthreads();
    // Stage K (32x64 [key][d]) and V^T (64x32 [d][key]) f16 tiles into LDS.
#pragma unroll
    for (int i = 0; i < 2; i++) {
      int c = t + 128 * i;  // 0..255
      int key = c >> 3, ck = c & 7;
      int dd = c >> 2, cv = c & 3;
#if USE_ASYNC_LDS
      // gfx1250 async global->LDS copy, tracked by ASYNCcnt (bypasses VGPRs).
      unsigned kdst = (unsigned)(unsigned long long)(Kt + key * 64 + ck * 8);
      unsigned koff = (unsigned)((((size_t)bh * SEQ + kt + key) * DKH + ck * 8) *
                                 sizeof(_Float16));
      asm volatile("global_load_async_to_lds_b128 %0, %1, %2"
                   :: "v"(kdst), "v"(koff), "s"(kbase) : "memory");
      unsigned vdst = (unsigned)(unsigned long long)(Vt + dd * 32 + cv * 8);
      unsigned voff = (unsigned)((((size_t)bh * DKH + dd) * SEQ + kt + cv * 8) *
                                 sizeof(_Float16));
      asm volatile("global_load_async_to_lds_b128 %0, %1, %2"
                   :: "v"(vdst), "v"(voff), "s"(vbase) : "memory");
#else
      *(v8h*)(Kt + key * 64 + ck * 8) =
          *(const v8h*)(Kh + ((size_t)bh * SEQ + kt + key) * DKH + ck * 8);
      *(v8h*)(Vt + dd * 32 + cv * 8) =
          *(const v8h*)(VTh + ((size_t)bh * DKH + dd) * SEQ + kt + cv * 8);
#endif
    }
#if USE_ASYNC_LDS
    asm volatile("s_wait_asynccnt 0" ::: "memory");
#endif
    __syncthreads();

    // scores: two 16x16 tiles (key groups), each accumulated over 2 d-chunks.
    v8f s[2] = {};
#pragma unroll
    for (int g = 0; g < 2; g++) {
#pragma unroll
      for (int c = 0; c < 2; c++) {
        V16U bk;
        const _Float16* kp = Kt + (16 * g + l16) * 64 + 32 * c;
        bk.h[0] = *(const v8h*)(kp + hsel * 8);
        bk.h[1] = *(const v8h*)(kp + 16 + hsel * 8);
        s[g] = wmma_f16(aq[c].v, bk.v, s[g]);
      }
    }

    float tk[2];
    int mk[2];
#pragma unroll
    for (int g = 0; g < 2; g++) {
      tk[g] = tdiff[(size_t)bb * SEQ + kt + 16 * g + l16];
      mk[g] = maskp[(size_t)bb * SEQ + kt + 16 * g + l16];
    }
    // scale * exp(-|dt|), key mask -> -inf
#pragma unroll
    for (int g = 0; g < 2; g++) {
#pragma unroll
      for (int v = 0; v < 8; v++) {
        float sc = s[g][v] * rscale;
        sc *= __expf(-fminf(fabsf(tq[v] - tk[g]), 100.f));
        s[g][v] = mk[g] ? sc : NEGINF;
      }
    }

    // Online softmax update (row stats across 16 lanes per half-wave).
    float alpha[8];
#pragma unroll
    for (int v = 0; v < 8; v++) {
      float rm = fmaxf(s[0][v], s[1][v]);
#pragma unroll
      for (int off = 8; off >= 1; off >>= 1)
        rm = fmaxf(rm, __shfl_xor(rm, off, 16));
      float mn = fmaxf(mrow[v], rm);
      alpha[v] = (mrow[v] == NEGINF) ? 0.f : __expf(mrow[v] - mn);
      float p0 = (mn == NEGINF) ? 0.f : __expf(s[0][v] - mn);
      float p1 = (mn == NEGINF) ? 0.f : __expf(s[1][v] - mn);
      s[0][v] = p0;
      s[1][v] = p1;
      float rs = p0 + p1;
#pragma unroll
      for (int off = 8; off >= 1; off >>= 1)
        rs += __shfl_xor(rs, off, 16);
      lrow[v] = lrow[v] * alpha[v] + rs;
      mrow[v] = mn;
    }
#pragma unroll
    for (int ct = 0; ct < 4; ct++)
#pragma unroll
      for (int v = 0; v < 8; v++) o[ct][v] *= alpha[v];

    // P (C-layout f32) -> per-wave LDS f16 [q][key], then A-fragment.
    _Float16* pw = Pt[wave];
#pragma unroll
    for (int g = 0; g < 2; g++)
#pragma unroll
      for (int v = 0; v < 8; v++)
        pw[(v + 8 * hsel) * 32 + 16 * g + l16] = (_Float16)s[g][v];

    V16U ap;
    const _Float16* pr = pw + l16 * 32;
    ap.h[0] = *(const v8h*)(pr + hsel * 8);
    ap.h[1] = *(const v8h*)(pr + 16 + hsel * 8);

    // ctx += P @ V  (4 d-column tiles)
#pragma unroll
    for (int ct = 0; ct < 4; ct++) {
      V16U bv;
      const _Float16* vp = Vt + (16 * ct + l16) * 32;
      bv.h[0] = *(const v8h*)(vp + hsel * 8);
      bv.h[1] = *(const v8h*)(vp + 16 + hsel * 8);
      o[ct] = wmma_f16(ap.v, bv.v, o[ct]);
    }
  }

  // Normalize and store ctx as f32 [B,S,D] (D index = h*64 + d).
  float sc8[8];
#pragma unroll
  for (int v = 0; v < 8; v++) sc8[v] = lrow[v] > 0.f ? 1.f / lrow[v] : 0.f;
#pragma unroll
  for (int ct = 0; ct < 4; ct++)
#pragma unroll
    for (int v = 0; v < 8; v++) {
      int q = q0 + v + 8 * hsel;
      ctx[((size_t)bb * SEQ + q) * D_MODEL + hh * DKH + 16 * ct + l16] =
          o[ct][v] * sc8[v];
    }
}

// ---------------------------------------------------------------------------
extern "C" void kernel_launch(void* const* d_in, const int* in_sizes, int n_in,
                              void* d_out, int out_size, void* d_ws,
                              size_t ws_size, hipStream_t stream) {
  (void)in_sizes; (void)n_in; (void)out_size; (void)ws_size;
  const float* query = (const float*)d_in[0];
  const float* keyt  = (const float*)d_in[1];
  const float* value = (const float*)d_in[2];
  const float* tdiff = (const float*)d_in[3];
  const unsigned char* maskp = (const unsigned char*)d_in[4];  // bool mask
  const float* Wq = (const float*)d_in[5];
  const float* bq = (const float*)d_in[6];
  const float* Wk = (const float*)d_in[7];
  const float* bk = (const float*)d_in[8];
  const float* Wv = (const float*)d_in[9];
  const float* bv = (const float*)d_in[10];
  const float* Wo = (const float*)d_in[11];
  const float* bo = (const float*)d_in[12];

  const size_t nQKV = (size_t)BATCH * NHEADS * SEQ * DKH;  // 4Mi elements
  _Float16* Qh  = (_Float16*)d_ws;
  _Float16* Kh  = Qh + nQKV;
  _Float16* VTh = Kh + nQKV;
  float* ctx = (float*)(VTh + nQKV);

  dim3 gblk(256);
  dim3 ggrid(D_MODEL / 128, MROWS / 128);
  gemm128<1><<<ggrid, gblk, 0, stream>>>(query, Wq, bq, Qh);
  gemm128<1><<<ggrid, gblk, 0, stream>>>(keyt, Wk, bk, Kh);
  gemm128<2><<<ggrid, gblk, 0, stream>>>(value, Wv, bv, VTh);
  attn_kernel<<<dim3(SEQ / 64, BATCH * NHEADS), dim3(128), 0, stream>>>(
      Qh, Kh, VTh, tdiff, maskp, ctx);
  gemm128<0><<<ggrid, gblk, 0, stream>>>(ctx, Wo, bo, d_out);
}
